// GCNModelVAE_FC_60601988546850
// MI455X (gfx1250) — compile-verified, matrix operand order checked by reading, not run on hardware
//
#include <hip/hip_runtime.h>
#include <hip/hip_bf16.h>

// ---------------------------------------------------------------------------
// GCN-VAE forward on gfx1250 (MI455X): bf16 WMMA GEMMs + atomic SpMM + fused
// bias/activation epilogues. Wave32, WGP, v_wmma_f32_16x16x32_bf16.
// GEMM: 128x128 block tile, 8 waves (4Mx2N), wave tile 32x64 (2x4 WMMA accs),
// double-buffered LDS, clamp-guarded edges (OOB rows/cols never stored).
// ---------------------------------------------------------------------------

#define N_   12000
#define FIN_ 3000
#define H_   512
#define Z_   64
#define HD_  512
#define KP1_ 3008   // FIN padded to multiple of 32

typedef __attribute__((ext_vector_type(16))) __bf16        v16bf;
typedef __attribute__((ext_vector_type(8)))  float         v8f;
typedef __attribute__((ext_vector_type(4)))  unsigned int  u32x4;

union Frag16 { u32x4 q[2]; v16bf v; };   // 16 bf16 = 32B = two b128 chunks
union U4x8   { u32x4 q;    unsigned short s[8]; };

__device__ __forceinline__ unsigned short f2bf(float f) {
  unsigned int u = __float_as_uint(f);
  unsigned int r = u + 0x7FFFu + ((u >> 16) & 1u);   // RNE
  return (unsigned short)(r >> 16);
}

// ------------------------------ cast kernels -------------------------------

__launch_bounds__(256)
__global__ void cast_pad_bf16(const float* __restrict__ src, unsigned short* __restrict__ dst,
                              int srcRows, int srcCols, int dstRows, int dstCols) {
  size_t total = (size_t)dstRows * dstCols;
  for (size_t i = (size_t)blockIdx.x * blockDim.x + threadIdx.x; i < total;
       i += (size_t)gridDim.x * blockDim.x) {
    int r = (int)(i / dstCols), c = (int)(i % dstCols);
    float v = (r < srcRows && c < srcCols) ? src[(size_t)r * srcCols + c] : 0.0f;
    dst[i] = f2bf(v);
  }
}

__launch_bounds__(256)
__global__ void cast_transpose_bf16(const float* __restrict__ src, unsigned short* __restrict__ dst,
                                    int srcRows, int srcCols) {
  // dst is [srcCols][srcRows]
  size_t total = (size_t)srcRows * srcCols;
  for (size_t i = (size_t)blockIdx.x * blockDim.x + threadIdx.x; i < total;
       i += (size_t)gridDim.x * blockDim.x) {
    int r = (int)(i / srcCols), c = (int)(i % srcCols);
    dst[(size_t)c * srcRows + r] = f2bf(src[i]);
  }
}

__launch_bounds__(256)
__global__ void zero_f32(float* __restrict__ p, size_t n) {
  for (size_t i = (size_t)blockIdx.x * blockDim.x + threadIdx.x; i < n;
       i += (size_t)gridDim.x * blockDim.x)
    p[i] = 0.0f;
}

// ------------------------------ WMMA GEMM ----------------------------------
// C[M,Nc] (f32) = A[M,Kp] (bf16 rowmajor) * B[Kp,Nc] (bf16 rowmajor)
// Requires: Kp % 32 == 0 (zero-padded in bf16 buffers), Nc % 8 == 0, Nc >= 8.

#define TM 128
#define TN 128
#define TK 32
#define LSTR 40   // LDS row stride in bf16 elems (80B) -> conflict-free b128

__launch_bounds__(256)
__global__ void wmma_gemm_bf16(const unsigned short* __restrict__ A,
                               const unsigned short* __restrict__ B,
                               float* __restrict__ C,
                               int M, int Nc, int Kp) {
  __shared__ __align__(16) unsigned short As[2][TM * LSTR];
  __shared__ __align__(16) unsigned short Bs[2][TN * LSTR];

  const int tid   = threadIdx.x;
  const int lane  = tid & 31;
  const int wave  = tid >> 5;
  const int waveM = wave >> 1;    // 0..3 -> 32 rows
  const int waveN = wave & 1;     // 0..1 -> 64 cols
  const int half  = lane >> 4;    // lanes 0-15 vs 16-31
  const int lr    = lane & 15;

  const int m0 = blockIdx.y * TM;
  const int n0 = blockIdx.x * TN;

  v8f acc[2][4] = {};

  // ---- staging thread mapping (clamped addresses; OOB tiles never stored) --
  // A: thread -> (row 0..127, 16-elem chunk at koff)
  const int ar    = tid >> 1;
  const int akoff = (tid & 1) * 16;
  int am = m0 + ar; if (am >= M) am = M - 1;
  const unsigned short* aBase = A + (size_t)am * Kp + akoff;
  // B: thread -> (k 0..31, two 8-col chunks at bng and bng+64), transposed
  const int bk  = tid & 31;
  const int bng = (tid >> 5) * 8;
  int bn0 = n0 + bng;      if (bn0 + 8 > Nc) bn0 = Nc - 8;
  int bn1 = n0 + bng + 64; if (bn1 + 8 > Nc) bn1 = Nc - 8;
  const unsigned short* bBase = B + (size_t)bk * Nc;

  const int ntiles = Kp / TK;

  // ---- prologue: stage tile 0 into buffer 0 ----
  u32x4 a0 = *(const u32x4*)(aBase);
  u32x4 a1 = *(const u32x4*)(aBase + 8);
  U4x8  bu0, bu1;
  bu0.q = *(const u32x4*)(bBase + bn0);
  bu1.q = *(const u32x4*)(bBase + bn1);

  *(u32x4*)&As[0][ar * LSTR + akoff]     = a0;
  *(u32x4*)&As[0][ar * LSTR + akoff + 8] = a1;
#pragma unroll
  for (int j = 0; j < 8; ++j) {
    Bs[0][(bng + j)      * LSTR + bk] = bu0.s[j];
    Bs[0][(bng + 64 + j) * LSTR + bk] = bu1.s[j];
  }
  __syncthreads();

  for (int t = 0; t < ntiles; ++t) {
    const int cur  = t & 1;
    const int nxt  = cur ^ 1;
    const bool more = (t + 1) < ntiles;

    // ---- issue next tile's global loads early (latency hides under WMMA) ---
    if (more) {
      const int kt = (t + 1) * TK;
      a0    = *(const u32x4*)(aBase + kt);
      a1    = *(const u32x4*)(aBase + kt + 8);
      bu0.q = *(const u32x4*)(bBase + (size_t)kt * Nc + bn0);
      bu1.q = *(const u32x4*)(bBase + (size_t)kt * Nc + bn1);
    }

    // ---- fragments from current buffer (ISA 16-bit layout) ----
    Frag16 fa[2], fb[4];
#pragma unroll
    for (int g = 0; g < 2; ++g) {
      int row = waveM * 32 + g * 16 + lr;
      fa[g].q[0] = *(const u32x4*)&As[cur][row * LSTR + half * 8];
      fa[g].q[1] = *(const u32x4*)&As[cur][row * LSTR + half * 8 + 16];
    }
#pragma unroll
    for (int h = 0; h < 4; ++h) {
      int col = waveN * 64 + h * 16 + lr;
      fb[h].q[0] = *(const u32x4*)&Bs[cur][col * LSTR + half * 8];
      fb[h].q[1] = *(const u32x4*)&Bs[cur][col * LSTR + half * 8 + 16];
    }

#pragma unroll
    for (int g = 0; g < 2; ++g)
#pragma unroll
      for (int h = 0; h < 4; ++h)
        acc[g][h] = __builtin_amdgcn_wmma_f32_16x16x32_bf16(
            false, fa[g].v, false, fb[h].v, (short)0, acc[g][h], false, false);

    // ---- commit next tile into the other buffer ----
    if (more) {
      *(u32x4*)&As[nxt][ar * LSTR + akoff]     = a0;
      *(u32x4*)&As[nxt][ar * LSTR + akoff + 8] = a1;
#pragma unroll
      for (int j = 0; j < 8; ++j) {
        Bs[nxt][(bng + j)      * LSTR + bk] = bu0.s[j];
        Bs[nxt][(bng + 64 + j) * LSTR + bk] = bu1.s[j];
      }
    }
    __syncthreads();
  }

  // ---- store C (accumulator layout: VGPR j -> row half*8+j, col lr) ----
#pragma unroll
  for (int g = 0; g < 2; ++g) {
#pragma unroll
    for (int h = 0; h < 4; ++h) {
      int cn = n0 + waveN * 64 + h * 16 + lr;
      if (cn >= Nc) continue;
#pragma unroll
      for (int j = 0; j < 8; ++j) {
        int cm = m0 + waveM * 32 + g * 16 + half * 8 + j;
        if (cm < M) C[(size_t)cm * Nc + cn] = acc[g][h][j];
      }
    }
  }
}

// ------------------------------- SpMM --------------------------------------
// out[row,:] += val * sup[col,:]  per edge, f32 global atomics.

__global__ void spmm_kernel(const int* __restrict__ rowi, const int* __restrict__ coli,
                            const float* __restrict__ val, const float* __restrict__ sup,
                            float* __restrict__ out, int D) {
  int e = blockIdx.x;
  int r = rowi[e], c = coli[e];
  float v = val[e];
  const float* s = sup + (size_t)c * D;
  float* o = out + (size_t)r * D;
  __builtin_prefetch(s, 0, 0);   // global_prefetch_b8
  for (int f = threadIdx.x; f < D; f += blockDim.x)
    atomicAdd(&o[f], v * s[f]);
}

// --------------------------- bias + activation -----------------------------
// mode: 1=leaky(0.01) 2=sigmoid 3=clip(softplus) 4=clip(exp)

__launch_bounds__(256)
__global__ void bias_act_kernel(const float* __restrict__ in, const float* __restrict__ bias,
                                float* __restrict__ out, size_t total, int D, int mode) {
  for (size_t i = (size_t)blockIdx.x * blockDim.x + threadIdx.x; i < total;
       i += (size_t)gridDim.x * blockDim.x) {
    float v = in[i] + bias[i % (size_t)D];
    float r;
    switch (mode) {
      case 1: r = (v < 0.0f) ? 0.01f * v : v; break;
      case 2: r = 1.0f / (1.0f + __expf(-v)); break;
      case 3: {
        float sp = (v > 20.0f) ? v : log1pf(__expf(v));
        r = fminf(fmaxf(sp, 1e-5f), 1e6f);
      } break;
      default: {
        float ex = __expf(fminf(v, 80.0f));
        r = fminf(fmaxf(ex, 1e-5f), 1e6f);
      } break;
    }
    out[i] = r;
  }
}

// ------------------------------- launcher ----------------------------------

extern "C" void kernel_launch(void* const* d_in, const int* in_sizes, int n_in,
                              void* d_out, int out_size, void* d_ws, size_t ws_size,
                              hipStream_t stream) {
  const int N = N_, FIN = FIN_, H = H_, Z = Z_, HD = HD_, KP1 = KP1_;
  const int E = in_sizes[1];

  const float* x       = (const float*)d_in[0];
  const int*   adj_row = (const int*)d_in[1];
  const int*   adj_col = (const int*)d_in[2];
  const float* adj_val = (const float*)d_in[3];
  const float* W1  = (const float*)d_in[4];  const float* b1  = (const float*)d_in[5];
  const float* W2  = (const float*)d_in[6];  const float* b2  = (const float*)d_in[7];
  const float* W2s = (const float*)d_in[8];  const float* b2s = (const float*)d_in[9];
  const float* Wd  = (const float*)d_in[10]; const float* bd  = (const float*)d_in[11];
  const float* Wpi = (const float*)d_in[12]; const float* bpi = (const float*)d_in[13];
  const float* Wth = (const float*)d_in[14]; const float* bth = (const float*)d_in[15];
  const float* Wmn = (const float*)d_in[16]; const float* bmn = (const float*)d_in[17];

  // ---- output layout (flat concat, reference return order) ----
  float* out    = (float*)d_out;
  float* A_rec  = out;
  float* mu     = A_rec + (size_t)N * N;
  float* logv   = mu    + (size_t)N * Z;
  float* zout   = logv  + (size_t)N * Z;
  float* dech   = zout  + (size_t)N * Z;
  float* pi     = dech  + (size_t)N * HD;
  float* theta  = pi    + (size_t)N * FIN;
  float* meanp  = theta + (size_t)N * FIN;

  // ---- workspace layout (region 0 reused: x_bf16 then decoder GEMM temp) ----
  char*  wsp = (char*)d_ws;
  size_t off = 0;
  auto take = [&](size_t bytes) -> void* {
    void* r = wsp + off;
    off += (bytes + 255) & ~(size_t)255;
    return r;
  };
  void* reg0 = take((size_t)N * FIN * 4);          // >= N*KP1*2 (bf16 x)
  unsigned short* xb   = (unsigned short*)reg0;    // phase 1
  float*          tbig = (float*)reg0;             // phase 2 (x dead)
  float*          sbig = (float*)take((size_t)N * FIN * 4);
  unsigned short* w1b  = (unsigned short*)take((size_t)KP1 * H * 2);
  float*          t512 = (float*)take((size_t)N * H * 4);
  float*          s512 = (float*)take((size_t)N * H * 4);
  unsigned short* hb   = (unsigned short*)take((size_t)N * H * 2);  // h1 then dec_h bf16
  unsigned short* w2b  = (unsigned short*)take((size_t)H * Z * 2);
  unsigned short* w2sb = (unsigned short*)take((size_t)H * Z * 2);
  unsigned short* wdb  = (unsigned short*)take((size_t)Z * HD * 2);
  unsigned short* wpib = (unsigned short*)take((size_t)HD * FIN * 2);
  unsigned short* wthb = (unsigned short*)take((size_t)HD * FIN * 2);
  unsigned short* wmnb = (unsigned short*)take((size_t)HD * FIN * 2);
  float*          t64  = (float*)take((size_t)N * Z * 4);
  float*          s64  = (float*)take((size_t)N * Z * 4);
  unsigned short* zb   = (unsigned short*)take((size_t)N * Z * 2);
  unsigned short* ztb  = (unsigned short*)take((size_t)Z * N * 2);
  (void)ws_size; (void)out_size; (void)n_in;

  // ---- helpers ----
  auto castPad = [&](const float* s, unsigned short* d, int sr, int sc, int dr, int dc) {
    size_t tot = (size_t)dr * dc;
    int blocks = (int)((tot + 255) / 256); if (blocks > 65535) blocks = 65535;
    cast_pad_bf16<<<blocks, 256, 0, stream>>>(s, d, sr, sc, dr, dc);
  };
  auto zero = [&](float* p, size_t n) {
    int blocks = (int)((n + 255) / 256); if (blocks > 65535) blocks = 65535;
    zero_f32<<<blocks, 256, 0, stream>>>(p, n);
  };
  auto gemm = [&](const unsigned short* A, const unsigned short* B, float* C,
                  int M, int Nc, int Kp) {
    dim3 g((Nc + TN - 1) / TN, (M + TM - 1) / TM);
    wmma_gemm_bf16<<<g, 256, 0, stream>>>(A, B, C, M, Nc, Kp);
  };
  auto spmm = [&](const float* sup, float* o, int D) {
    int thr = (D >= 256) ? 256 : ((D >= 64) ? 64 : 32);
    spmm_kernel<<<E, thr, 0, stream>>>(adj_row, adj_col, adj_val, sup, o, D);
  };
  auto biasAct = [&](const float* in, const float* b, float* o, int M, int D, int mode) {
    size_t tot = (size_t)M * D;
    int blocks = (int)((tot + 255) / 256); if (blocks > 65535) blocks = 65535;
    bias_act_kernel<<<blocks, 256, 0, stream>>>(in, b, o, tot, D, mode);
  };

  // ---- weight / input casts ----
  castPad(x,   xb,   N,   FIN, N,   KP1);
  castPad(W1,  w1b,  FIN, H,   KP1, H);
  castPad(W2,  w2b,  H,   Z,   H,   Z);
  castPad(W2s, w2sb, H,   Z,   H,   Z);
  castPad(Wd,  wdb,  Z,   HD,  Z,   HD);
  castPad(Wpi, wpib, HD,  FIN, HD,  FIN);
  castPad(Wth, wthb, HD,  FIN, HD,  FIN);
  castPad(Wmn, wmnb, HD,  FIN, HD,  FIN);

  // ---- encoder layer 1: h1 = leaky(spmm(x@W1) + b1) ----
  gemm(xb, w1b, t512, N, H, KP1);
  zero(s512, (size_t)N * H);
  spmm(t512, s512, H);
  biasAct(s512, b1, s512, N, H, 1);
  castPad(s512, hb, N, H, N, H);

  // ---- mu ----
  gemm(hb, w2b, t64, N, Z, H);
  zero(s64, (size_t)N * Z);
  spmm(t64, s64, Z);
  biasAct(s64, b2, mu, N, Z, 1);

  // ---- logvar ----
  gemm(hb, w2sb, t64, N, Z, H);
  zero(s64, (size_t)N * Z);
  spmm(t64, s64, Z);
  biasAct(s64, b2s, logv, N, Z, 1);

  // ---- z = mu ----
  hipMemcpyAsync(zout, mu, (size_t)N * Z * sizeof(float),
                 hipMemcpyDeviceToDevice, stream);

  // ---- A_rec = z @ z^T ----
  castPad(mu, zb, N, Z, N, Z);
  cast_transpose_bf16<<<3000, 256, 0, stream>>>(mu, ztb, N, Z);
  gemm(zb, ztb, A_rec, N, N, Z);

  // ---- dec_h = leaky(spmm(z@Wd) + bd) ----
  gemm(zb, wdb, t512, N, HD, Z);
  zero(s512, (size_t)N * HD);
  spmm(t512, s512, HD);
  biasAct(s512, bd, dech, N, HD, 1);
  castPad(dech, hb, N, HD, N, HD);

  // ---- ZINB heads ----
  // pi = sigmoid(spmm(dec_h@Wpi) + bpi)
  gemm(hb, wpib, tbig, N, FIN, HD);
  zero(sbig, (size_t)N * FIN);
  spmm(tbig, sbig, FIN);
  biasAct(sbig, bpi, pi, N, FIN, 2);
  // theta = clip(softplus(spmm(dec_h@Wth) + bth))
  gemm(hb, wthb, tbig, N, FIN, HD);
  zero(sbig, (size_t)N * FIN);
  spmm(tbig, sbig, FIN);
  biasAct(sbig, bth, theta, N, FIN, 3);
  // mean = clip(exp(spmm(dec_h@Wmn) + bmn))
  gemm(hb, wmnb, tbig, N, FIN, HD);
  zero(sbig, (size_t)N * FIN);
  spmm(tbig, sbig, FIN);
  biasAct(sbig, bmn, meanp, N, FIN, 4);
}
